// NCELoss_39548058861731
// MI455X (gfx1250) — matrix operand order repaired and checked by reading, not compiled, not run on hardware
//
#include <hip/hip_runtime.h>
#include <hip/hip_bf16.h>

// ---------------------------------------------------------------------------
// SimCLR NT-Xent loss, fused for MI455X (gfx1250, wave32, WMMA).
//   B=4096, D=256, n=2B=8192, TEMP=0.5  =>  sim scale = 1/TEMP = 2.0
// sim is never materialized: GEMM tiles feed an exp-sum with a FIXED max of
// 2.0 (cosine/temp <= 2), so partial sums over column chunks combine linearly
// and the j dimension splits across blocks. The diagonal term exp(sim_ii - 2)
// is subtracted analytically in the per-row pass (sim_ii = 2*||zhat_i||^2),
// keeping the WMMA inner loop mask-free.
// LDS panel rows are padded to 528 B so the 16 fragment-loading lanes of each
// half-wave hit 16 distinct 4-bank groups (conflict-free ds_load_b128).
// ---------------------------------------------------------------------------

#define BATCH   4096
#define DIM     256
#define NROWS   8192          // 2*BATCH
#define INV_T   2.0f          // 1/TEMP
#define JSPLIT  8             // column chunks (blocks in y)
#define PANEL   128           // j-rows staged in LDS per panel
#define LDSSTR  264           // padded LDS row stride in halves (528 B):
                              // 528 % 256 == 16 -> conflict-free b128 reads
#define K2LOG2E 2.8853900817779268f   // 2 * log2(e); exp(2c-2)=exp2(c*K - K)

typedef __attribute__((ext_vector_type(16))) _Float16 v16h;
typedef __attribute__((ext_vector_type(8)))  _Float16 v8h_t;
typedef __attribute__((ext_vector_type(8)))  float    v8f;

union FragAB { v16h v; v8h_t h[2]; };

// ---------------------------------------------------------------------------
// Kernel 1: z = concat(emb1, emb2); zn[i] = f16(z[i] / ||z[i]||)
// One wave per row; 8 floats per lane; shuffle reduction for sum of squares.
// ---------------------------------------------------------------------------
__global__ __launch_bounds__(256)
void nce_normalize(const float* __restrict__ emb1,
                   const float* __restrict__ emb2,
                   _Float16* __restrict__ zn) {
    const int lane = threadIdx.x & 31;
    const int wave = threadIdx.x >> 5;
    const int row  = blockIdx.x * 8 + wave;

    const float* src = (row < BATCH) ? (emb1 + (size_t)row * DIM)
                                     : (emb2 + (size_t)(row - BATCH) * DIM);
    const float4* s4 = (const float4*)(src + lane * 8);
    float4 x0 = s4[0];
    float4 x1 = s4[1];

    float ss = x0.x*x0.x + x0.y*x0.y + x0.z*x0.z + x0.w*x0.w
             + x1.x*x1.x + x1.y*x1.y + x1.z*x1.z + x1.w*x1.w;
    #pragma unroll
    for (int off = 16; off >= 1; off >>= 1)
        ss += __shfl_xor(ss, off, 32);

    const float inv = 1.0f / sqrtf(ss);   // norms ~16 for N(0,1) data; eps moot

    v8h_t hv;
    hv[0] = (_Float16)(x0.x * inv); hv[1] = (_Float16)(x0.y * inv);
    hv[2] = (_Float16)(x0.z * inv); hv[3] = (_Float16)(x0.w * inv);
    hv[4] = (_Float16)(x1.x * inv); hv[5] = (_Float16)(x1.y * inv);
    hv[6] = (_Float16)(x1.z * inv); hv[7] = (_Float16)(x1.w * inv);
    *(v8h_t*)(zn + (size_t)row * DIM + lane * 8) = hv;
}

// ---------------------------------------------------------------------------
// Kernel 2: fused sim GEMM + exp-sum (fixed max = 2.0, diagonal INCLUDED).
// grid = (NROWS/128, JSPLIT); block = 256 (8 waves, 16 rows each).
// A fragments (16 rows x K=256) live in registers for the whole column sweep.
// B panels (128 j-rows x 256) staged in LDS (padded stride); two 16-col tiles
// per iteration give two independent WMMA accumulation chains.
// Output: sumpart[jc*NROWS + i] = sum_{j in chunk} exp(sim[i][j] - 2)
// ---------------------------------------------------------------------------
__global__ __launch_bounds__(256)
void nce_simpanel(const _Float16* __restrict__ zn,
                  float* __restrict__ sumpart) {
    __shared__ _Float16 ldsB[PANEL * LDSSTR];   // 66 KB, padded rows

    const int lane   = threadIdx.x & 31;
    const int wave   = threadIdx.x >> 5;
    const int lm     = lane & 15;               // N (cols) / M (rows) selector
    const int laneHi = lane >> 4;               // 0: lanes 0-15, 1: lanes 16-31
    const int i0     = blockIdx.x * 128 + wave * 16;
    const int jc     = blockIdx.y;

    // --- preload A fragments: lane holds row i0+lm, contiguous K chunks (same order as B)
    //     lanes 0-15 : K = {32t..32t+15}
    //     lanes 16-31: K = {32t+16..32t+31}
    FragAB a[8];
    const _Float16* zrow = zn + (size_t)(i0 + lm) * DIM;
    #pragma unroll
    for (int t = 0; t < 8; ++t) {
        const int c0 = 32 * t + 16 * laneHi;
        a[t].h[0] = *(const v8h_t*)(zrow + c0);
        a[t].h[1] = *(const v8h_t*)(zrow + c0 + 8);
    }

    float sumr[8];
    #pragma unroll
    for (int r = 0; r < 8; ++r) sumr[r] = 0.0f;

    const int jbeg = jc * (NROWS / JSPLIT);
    const int jend = jbeg + (NROWS / JSPLIT);

    for (int p0 = jbeg; p0 < jend; p0 += PANEL) {
        __syncthreads();                         // guard LDS reuse
        {   // cooperative panel copy: 4096 float4; wave q%32 lanes cover one
            // row (32 consecutive float4) -> conflict-free ds_store_b128
            const float4* src = (const float4*)(zn + (size_t)p0 * DIM);
            #pragma unroll
            for (int q = 0; q < 16; ++q) {
                const int idx = threadIdx.x + 256 * q;
                const int row = idx >> 5;        // 0..127
                const int col = idx & 31;        // 0..31 (float4 within row)
                ((float4*)(ldsB + row * LDSSTR))[col] = src[idx];
            }
        }
        __syncthreads();

        #pragma unroll 1
        for (int tt = 0; tt < PANEL / 16; tt += 2) {
            // B fragment: lane holds column j0+lm (= zn row j0+lm),
            // K = 32t + 16*laneHi .. +15 contiguous.
            const _Float16* brow0 = ldsB + (size_t)(tt * 16 + lm) * LDSSTR;
            const _Float16* brow1 = brow0 + 16 * LDSSTR;
            v8f c0 = {};
            v8f c1 = {};
            #pragma unroll
            for (int t = 0; t < 8; ++t) {
                const int k0 = 32 * t + 16 * laneHi;
                FragAB b0, b1;
                b0.h[0] = *(const v8h_t*)(brow0 + k0);
                b0.h[1] = *(const v8h_t*)(brow0 + k0 + 8);
                b1.h[0] = *(const v8h_t*)(brow1 + k0);
                b1.h[1] = *(const v8h_t*)(brow1 + k0 + 8);
                c0 = __builtin_amdgcn_wmma_f32_16x16x32_f16(
                        false, a[t].v, false, b0.v, (short)0, c0, false, false);
                c1 = __builtin_amdgcn_wmma_f32_16x16x32_f16(
                        false, a[t].v, false, b1.v, (short)0, c1, false, false);
            }
            // exp(sim - 2) = exp2(c * 2log2e - 2log2e): one FMA + one v_exp
            #pragma unroll
            for (int r = 0; r < 8; ++r) {
                sumr[r] += __builtin_amdgcn_exp2f(fmaf(c0[r], K2LOG2E, -K2LOG2E))
                         + __builtin_amdgcn_exp2f(fmaf(c1[r], K2LOG2E, -K2LOG2E));
            }
        }
    }

    // combine the 16 column-lanes of each half-wave (rows identical per half)
    #pragma unroll
    for (int off = 1; off <= 8; off <<= 1)
        #pragma unroll
        for (int r = 0; r < 8; ++r)
            sumr[r] += __shfl_xor(sumr[r], off, 32);

    if (lm == 0) {   // lane 0 -> rows i0..i0+7 ; lane 16 -> rows i0+8..i0+15
        const int base = i0 + 8 * laneHi;
        #pragma unroll
        for (int r = 0; r < 8; ++r)
            sumpart[(size_t)jc * NROWS + base + r] = sumr[r];
    }
}

// ---------------------------------------------------------------------------
// Kernel 3: per-row term = lse_i - pos_i
//   lse_i = 2 + log( sum_jc sumpart[jc][i] - exp(2*||zn_i||^2 - 2) )
//   pos_i = 2 * dot(zn[i], zn[(i+B) mod n])
// One wave per row.
// ---------------------------------------------------------------------------
__global__ __launch_bounds__(256)
void nce_rowterm(const _Float16* __restrict__ zn,
                 const float* __restrict__ sumpart,
                 float* __restrict__ rowterm) {
    const int lane = threadIdx.x & 31;
    const int wave = threadIdx.x >> 5;
    const int i    = blockIdx.x * 8 + wave;
    const int j    = (i < BATCH) ? (i + BATCH) : (i - BATCH);

    v8h_t xa = *(const v8h_t*)(zn + (size_t)i * DIM + lane * 8);
    v8h_t xb = *(const v8h_t*)(zn + (size_t)j * DIM + lane * 8);
    float dot  = 0.0f;   // positive-pair logit (pre-scale)
    float self = 0.0f;   // ||zhat_i||^2 for diagonal removal
    #pragma unroll
    for (int e = 0; e < 8; ++e) {
        const float fa = (float)xa[e];
        dot  = fmaf(fa, (float)xb[e], dot);
        self = fmaf(fa, fa, self);
    }
    #pragma unroll
    for (int off = 16; off >= 1; off >>= 1) {
        dot  += __shfl_xor(dot,  off, 32);
        self += __shfl_xor(self, off, 32);
    }

    if (lane == 0) {
        float s = 0.0f;
        #pragma unroll
        for (int c = 0; c < JSPLIT; ++c)
            s += sumpart[(size_t)c * NROWS + i];
        s -= __builtin_amdgcn_exp2f(fmaf(self, K2LOG2E, -K2LOG2E));  // drop diag
        rowterm[i] = (2.0f + logf(s)) - dot * INV_T;
    }
}

// ---------------------------------------------------------------------------
// Kernel 4: deterministic tree reduction of 8192 row terms -> loss
// ---------------------------------------------------------------------------
__global__ __launch_bounds__(256)
void nce_reduce(const float* __restrict__ rowterm, float* __restrict__ out) {
    __shared__ float sh[256];
    float s = 0.0f;
    for (int i = threadIdx.x; i < NROWS; i += 256)
        s += rowterm[i];
    sh[threadIdx.x] = s;
    __syncthreads();
    for (int off = 128; off > 0; off >>= 1) {
        if (threadIdx.x < off) sh[threadIdx.x] += sh[threadIdx.x + off];
        __syncthreads();
    }
    if (threadIdx.x == 0) out[0] = sh[0] / (float)NROWS;
}

// ---------------------------------------------------------------------------
extern "C" void kernel_launch(void* const* d_in, const int* in_sizes, int n_in,
                              void* d_out, int out_size, void* d_ws, size_t ws_size,
                              hipStream_t stream) {
    const float* emb1 = (const float*)d_in[0];
    const float* emb2 = (const float*)d_in[1];
    float*       out  = (float*)d_out;

    // workspace layout
    _Float16* zn      = (_Float16*)d_ws;                                 // 4 MB
    float*    sumpart = (float*)((char*)d_ws + (size_t)NROWS * DIM * 2); // 256 KB
    float*    rowterm = sumpart + (size_t)JSPLIT * NROWS;                // 32 KB

    nce_normalize<<<NROWS / 8, 256, 0, stream>>>(emb1, emb2, zn);
    nce_simpanel<<<dim3(NROWS / 128, JSPLIT), 256, 0, stream>>>(zn, sumpart);
    nce_rowterm<<<NROWS / 8, 256, 0, stream>>>(zn, sumpart, rowterm);
    nce_reduce<<<1, 256, 0, stream>>>(rowterm, out);
}
